// GNSModel_88098369175668
// MI455X (gfx1250) — matrix operand order repaired and checked by reading, not compiled, not run on hardware
//
#include <hip/hip_runtime.h>
#include <math.h>

typedef __attribute__((ext_vector_type(16))) _Float16 v16h;
typedef __attribute__((ext_vector_type(8)))  float    v8f;

// ---------------------------------------------------------------------------
// Async global->LDS copy (CDNA5): GVS mode, 16 bytes per lane.
//   VDST vgpr = per-lane LDS byte offset, VADDR vgpr = 32-bit byte offset,
//   SADDR = uniform 64-bit base.  Tracked by ASYNCcnt.
// ---------------------------------------------------------------------------
__device__ inline void async_load_b128(const void* saddr, unsigned voff, unsigned ldsoff)
{
    asm volatile("global_load_async_to_lds_b128 %0, %1, %2"
                 :: "v"(ldsoff), "v"(voff), "s"(saddr)
                 : "memory");
}
__device__ inline void wait_async0()
{
    asm volatile("s_wait_asynccnt 0x0" ::: "memory");
}

// ---------------------------------------------------------------------------
// Per-wave 16x16 output-tile GEMM: C += A[16 x 32*ktiles] * B[32*ktiles x 16]
// A staged in LDS (f16 row-major, 'astride' halves/row, even stride).
// B pre-swizzled in global memory in WMMA fragment order, kb-major:
//   frag entry index = kb*32 + lane, each entry = v16h.
// A fragment (ISA 7.12.2, 16-bit A 16x32): lane L: M=L&15;
//   lanes 0-15 hold K 0..7,16..23; lanes 16-31 hold K 8..15,24..31.
// ---------------------------------------------------------------------------
__device__ inline v8f wmma_gemm_lds(const _Float16* __restrict__ A, int astride,
                                    const v16h* __restrict__ Bfrag, int ktiles,
                                    int lane)
{
    v8f c = {};
    const _Float16* arow = A + (lane & 15) * astride + ((lane & 16) ? 8 : 0);
    for (int kb = 0; kb < ktiles; ++kb) {
        union { unsigned u[8]; v16h h; } av;
        const _Float16* ap = arow + kb * 32;
#pragma unroll
        for (int p = 0; p < 4; ++p) {
            av.u[p]     = *(const unsigned*)(ap + 2 * p);
            av.u[p + 4] = *(const unsigned*)(ap + 16 + 2 * p);
        }
        v16h b = Bfrag[kb * 32 + lane];
        c = __builtin_amdgcn_wmma_f32_16x16x32_f16(false, av.h, false, b,
                                                   (short)0, c, false, false);
    }
    return c;
}

// Two row-tiles sharing each B fragment (halves weight traffic, 2 wmma / B load)
__device__ inline void wmma_gemm_lds_x2(const _Float16* __restrict__ A, int astride,
                                        const v16h* __restrict__ Bfrag, int ktiles,
                                        int lane, v8f& c0, v8f& c1)
{
    const _Float16* arow0 = A + (lane & 15) * astride + ((lane & 16) ? 8 : 0);
    const _Float16* arow1 = arow0 + 16 * astride;
    for (int kb = 0; kb < ktiles; ++kb) {
        v16h b = Bfrag[kb * 32 + lane];
        union { unsigned u[8]; v16h h; } a0, a1;
        const _Float16* ap0 = arow0 + kb * 32;
        const _Float16* ap1 = arow1 + kb * 32;
#pragma unroll
        for (int p = 0; p < 4; ++p) {
            a0.u[p]     = *(const unsigned*)(ap0 + 2 * p);
            a0.u[p + 4] = *(const unsigned*)(ap0 + 16 + 2 * p);
            a1.u[p]     = *(const unsigned*)(ap1 + 2 * p);
            a1.u[p + 4] = *(const unsigned*)(ap1 + 16 + 2 * p);
        }
        c0 = __builtin_amdgcn_wmma_f32_16x16x32_f16(false, a0.h, false, b,
                                                    (short)0, c0, false, false);
        c1 = __builtin_amdgcn_wmma_f32_16x16x32_f16(false, a1.h, false, b,
                                                    (short)0, c1, false, false);
    }
}

// C-frag: VGPR r -> row = r + (lane&16 ? 8 : 0), col = lane&15
__device__ inline void store_hidden_relu(v8f c, const float* __restrict__ bias,
                                         _Float16* H, int hstride, int ntile, int lane)
{
    const int col = ntile * 16 + (lane & 15);
    const int rb  = (lane & 16) ? 8 : 0;
    const float b = bias[col];
#pragma unroll
    for (int r = 0; r < 8; ++r) {
        float v = c[r] + b;
        H[(rb + r) * hstride + col] = (_Float16)(v > 0.0f ? v : 0.0f);
    }
}

// ---------------------------------------------------------------------------
// f32 weight [Ksrc x Ncols] -> f16 WMMA B-fragment layout (K zero-padded)
// ---------------------------------------------------------------------------
__global__ void k_convert_weight(const float* __restrict__ w, _Float16* __restrict__ out,
                                 int Ksrc, int KB, int NB, int Ncols)
{
    const int total = NB * KB * 512;
    for (int idx = blockIdx.x * blockDim.x + threadIdx.x; idx < total;
         idx += gridDim.x * blockDim.x) {
        const int h    = idx & 15;
        const int lane = (idx >> 4) & 31;
        const int blk  = idx >> 9;
        const int kb   = blk % KB;
        const int nb   = blk / KB;
        const int n = nb * 16 + (lane & 15);
        const int k = kb * 32 + ((lane & 16) ? 8 : 0) + (h < 8 ? h : h + 8);
        const float v = (k < Ksrc) ? w[(size_t)k * Ncols + n] : 0.0f;
        out[idx] = (_Float16)v;
    }
}

__global__ void k_zero(float* __restrict__ p, long n)
{
    long i = (long)blockIdx.x * blockDim.x + threadIdx.x;
    const long stride = (long)gridDim.x * blockDim.x;
    for (; i < n; i += stride) p[i] = 0.0f;
}

__global__ void k_degree(const int* __restrict__ recv, float* __restrict__ deg, int E)
{
    int i = blockIdx.x * blockDim.x + threadIdx.x;
    const int stride = gridDim.x * blockDim.x;
    for (; i < E; i += stride) atomicAdd(&deg[recv[i]], 1.0f);
}

__global__ __launch_bounds__(256) void k_center(const float* __restrict__ pos,
                                                float* __restrict__ csum, int N)
{
    float sx = 0.f, sy = 0.f, sz = 0.f;
    int i = blockIdx.x * blockDim.x + threadIdx.x;
    const int stride = gridDim.x * blockDim.x;
    for (; i < N; i += stride) {
        sx += pos[(size_t)i * 3 + 0];
        sy += pos[(size_t)i * 3 + 1];
        sz += pos[(size_t)i * 3 + 2];
    }
    __shared__ float sh[3 * 256];
    sh[threadIdx.x] = sx; sh[256 + threadIdx.x] = sy; sh[512 + threadIdx.x] = sz;
    __syncthreads();
    for (int off = 128; off > 0; off >>= 1) {
        if ((int)threadIdx.x < off) {
            sh[threadIdx.x]       += sh[threadIdx.x + off];
            sh[256 + threadIdx.x] += sh[256 + threadIdx.x + off];
            sh[512 + threadIdx.x] += sh[512 + threadIdx.x + off];
        }
        __syncthreads();
    }
    if (threadIdx.x == 0) {
        atomicAdd(&csum[0], sh[0]);
        atomicAdd(&csum[1], sh[256]);
        atomicAdd(&csum[2], sh[512]);
    }
}

// ---------------------------------------------------------------------------
// Node encoder: features [16 nodes x 23 (pad 32)] -> L -> L
// ---------------------------------------------------------------------------
__global__ __launch_bounds__(256) void k_node_encoder(
    const float* __restrict__ pos, const float* __restrict__ vel,
    const int* __restrict__ rid, const float* __restrict__ remb,
    const float* __restrict__ deg, const float* __restrict__ csum,
    float* __restrict__ nlf, _Float16* __restrict__ nl16,
    const v16h* __restrict__ W1, const float* __restrict__ B1,
    const v16h* __restrict__ W2, const float* __restrict__ B2, float invN)
{
    __shared__ _Float16 Ain[16 * 40];
    __shared__ _Float16 Hid[16 * 136];
    const int tid = threadIdx.x, lane = tid & 31, wave = tid >> 5;
    const long nb0 = (long)blockIdx.x * 16;

    for (int idx = tid; idx < 16 * 32; idx += 256) {
        const int row = idx >> 5, c = idx & 31;
        const long n = nb0 + row;
        float v = 0.0f;
        if (c < 3)        v = vel[n * 3 + c];
        else if (c < 6)   v = pos[n * 3 + (c - 3)] - csum[c - 3] * invN;
        else if (c == 6)  v = deg[n];
        else if (c < 23)  v = remb[(size_t)rid[n] * 16 + (c - 7)];
        Ain[row * 40 + c] = (_Float16)v;
    }
    __syncthreads();
    v8f h = wmma_gemm_lds(Ain, 40, W1 + (size_t)wave * 32, 1, lane);
    store_hidden_relu(h, B1, Hid, 136, wave, lane);
    __syncthreads();
    v8f o = wmma_gemm_lds(Hid, 136, W2 + (size_t)wave * 4 * 32, 4, lane);
    const int col = wave * 16 + (lane & 15);
    const int rb  = (lane & 16) ? 8 : 0;
    const float b2 = B2[col];
#pragma unroll
    for (int r = 0; r < 8; ++r) {
        const long n = nb0 + rb + r;
        const float v = o[r] + b2;
        nlf[n * 128 + col]  = v;
        nl16[n * 128 + col] = (_Float16)v;
    }
}

// ---------------------------------------------------------------------------
// Edge encoder: features [16 edges x 9 (pad 32)] -> L -> L
// ---------------------------------------------------------------------------
__global__ __launch_bounds__(256) void k_edge_encoder(
    const float* __restrict__ pos, const float* __restrict__ vel,
    const int* __restrict__ rid,
    const int* __restrict__ snd, const int* __restrict__ rcv,
    _Float16* __restrict__ el16,
    const v16h* __restrict__ W1, const float* __restrict__ B1,
    const v16h* __restrict__ W2, const float* __restrict__ B2)
{
    __shared__ _Float16 Ain[16 * 40];
    __shared__ _Float16 Hid[16 * 136];
    const int tid = threadIdx.x, lane = tid & 31, wave = tid >> 5;
    const long eb = (long)blockIdx.x * 16;

    if (tid < 16) {
        const long e = eb + tid;
        const int s = snd[e], r = rcv[e];
        const float rp0 = pos[(size_t)s * 3 + 0] - pos[(size_t)r * 3 + 0];
        const float rp1 = pos[(size_t)s * 3 + 1] - pos[(size_t)r * 3 + 1];
        const float rp2 = pos[(size_t)s * 3 + 2] - pos[(size_t)r * 3 + 2];
        const float rv0 = vel[(size_t)s * 3 + 0] - vel[(size_t)r * 3 + 0];
        const float rv1 = vel[(size_t)s * 3 + 1] - vel[(size_t)r * 3 + 1];
        const float rv2 = vel[(size_t)s * 3 + 2] - vel[(size_t)r * 3 + 2];
        const float sq = rp0 * rp0 + rp1 * rp1 + rp2 * rp2;
        const float dist = sqrtf(sq);
        const float same = (rid[s] == rid[r]) ? 1.0f : 0.0f;
        _Float16* a = Ain + tid * 40;
        a[0] = (_Float16)rp0; a[1] = (_Float16)rp1; a[2] = (_Float16)rp2;
        a[3] = (_Float16)rv0; a[4] = (_Float16)rv1; a[5] = (_Float16)rv2;
        a[6] = (_Float16)dist; a[7] = (_Float16)sq; a[8] = (_Float16)same;
#pragma unroll
        for (int c = 9; c < 32; ++c) a[c] = (_Float16)0.0f;
    }
    __syncthreads();
    v8f h = wmma_gemm_lds(Ain, 40, W1 + (size_t)wave * 32, 1, lane);
    store_hidden_relu(h, B1, Hid, 136, wave, lane);
    __syncthreads();
    v8f o = wmma_gemm_lds(Hid, 136, W2 + (size_t)wave * 4 * 32, 4, lane);
    const int col = wave * 16 + (lane & 15);
    const int rb  = (lane & 16) ? 8 : 0;
    const float b2 = B2[col];
#pragma unroll
    for (int r = 0; r < 8; ++r) {
        const long e = eb + rb + r;
        el16[e * 128 + col] = (_Float16)(o[r] + b2);
    }
}

// ---------------------------------------------------------------------------
// Processor edge MLP (dominant kernel): 32 edges/block, 8 waves, each wave
// computes a 32x16 output slab (two 16x16 row-tiles sharing B fragments).
// Staging via async global->LDS b128 (ASYNCcnt).  Fused residual + scatter.
//   in  = [el | nl[s] | nl[r]]  (32 x 384);  el += MLP(in);  agg[rcv] += el
// ---------------------------------------------------------------------------
__global__ __launch_bounds__(256) void k_edge_mlp(
    const _Float16* __restrict__ nl16, _Float16* __restrict__ el16,
    float* __restrict__ agg,
    const int* __restrict__ snd, const int* __restrict__ rcv,
    const v16h* __restrict__ W1, const float* __restrict__ B1,
    const v16h* __restrict__ W2, const float* __restrict__ B2)
{
    __shared__ __align__(16) _Float16 Ain[32 * 392];   // 384 + 8 pad (row = 784 B)
    __shared__ __align__(16) _Float16 Hid[32 * 136];
    __shared__ int sSend[32], sRecv[32];
    const int tid = threadIdx.x, lane = tid & 31, wave = tid >> 5;
    const long eb = (long)blockIdx.x * 32;

    if (tid < 32) {
        sSend[tid] = snd[eb + tid];
        sRecv[tid] = rcv[eb + tid];
    }
    __syncthreads();

    const unsigned ainB = (unsigned)(uintptr_t)&Ain[0];   // LDS byte offset
    // seg 0: el rows (row-major f16, 256 B per row = 16 chunks of 16 B)
    for (int idx = tid; idx < 512; idx += 256) {
        const int row = idx >> 4, ch = idx & 15;
        async_load_b128(el16, (unsigned)((eb + row) * 256) + ch * 16,
                        ainB + row * 784 + ch * 16);
    }
    // seg 1: nl[senders]
    for (int idx = tid; idx < 512; idx += 256) {
        const int row = idx >> 4, ch = idx & 15;
        async_load_b128(nl16, (unsigned)(sSend[row] * 256) + ch * 16,
                        ainB + row * 784 + 256 + ch * 16);
    }
    // seg 2: nl[receivers]
    for (int idx = tid; idx < 512; idx += 256) {
        const int row = idx >> 4, ch = idx & 15;
        async_load_b128(nl16, (unsigned)(sRecv[row] * 256) + ch * 16,
                        ainB + row * 784 + 512 + ch * 16);
    }
    wait_async0();
    __syncthreads();

    v8f h0 = {}, h1 = {};
    wmma_gemm_lds_x2(Ain, 392, W1 + (size_t)wave * 12 * 32, 12, lane, h0, h1);
    store_hidden_relu(h0, B1, Hid, 136, wave, lane);
    store_hidden_relu(h1, B1, Hid + 16 * 136, 136, wave, lane);
    __syncthreads();

    v8f o0 = {}, o1 = {};
    wmma_gemm_lds_x2(Hid, 136, W2 + (size_t)wave * 4 * 32, 4, lane, o0, o1);

    const int col = wave * 16 + (lane & 15);
    const int rb  = (lane & 16) ? 8 : 0;
    const float b2 = B2[col];
    v8f outs[2] = { o0, o1 };
#pragma unroll
    for (int rt = 0; rt < 2; ++rt) {
#pragma unroll
        for (int r = 0; r < 8; ++r) {
            const int row = rt * 16 + rb + r;
            const long e = eb + row;
            const float old = (float)Ain[row * 392 + col];   // seg0 = old el
            const float v = outs[rt][r] + b2 + old;
            el16[e * 128 + col] = (_Float16)v;
            atomicAdd(&agg[(size_t)sRecv[row] * 128 + col], v);
        }
    }
}

// ---------------------------------------------------------------------------
// Processor node MLP + residual: in = [nl | agg] (16 nodes x 256), nl += MLP
// ---------------------------------------------------------------------------
__global__ __launch_bounds__(256) void k_node_mlp(
    float* __restrict__ nlf, _Float16* __restrict__ nl16,
    const float* __restrict__ agg,
    const v16h* __restrict__ W1, const float* __restrict__ B1,
    const v16h* __restrict__ W2, const float* __restrict__ B2)
{
    __shared__ _Float16 Ain[16 * 264];   // 256 + 8 pad
    __shared__ _Float16 Hid[16 * 136];
    const int tid = threadIdx.x, lane = tid & 31, wave = tid >> 5;
    const long nb0 = (long)blockIdx.x * 16;

    for (int idx = tid; idx < 16 * 64; idx += 256) {     // seg0: nl16 dword pairs
        const int row = idx >> 6, cp = idx & 63;
        const unsigned* src = (const unsigned*)(nl16 + (nb0 + row) * 128);
        *(unsigned*)(Ain + row * 264 + cp * 2) = src[cp];
    }
    for (int idx = tid; idx < 16 * 128; idx += 256) {    // seg1: agg f32 -> f16
        const int row = idx >> 7, c = idx & 127;
        Ain[row * 264 + 128 + c] = (_Float16)agg[(nb0 + row) * 128 + c];
    }
    __syncthreads();

    v8f h = wmma_gemm_lds(Ain, 264, W1 + (size_t)wave * 8 * 32, 8, lane);
    store_hidden_relu(h, B1, Hid, 136, wave, lane);
    __syncthreads();
    v8f o = wmma_gemm_lds(Hid, 136, W2 + (size_t)wave * 4 * 32, 4, lane);

    const int col = wave * 16 + (lane & 15);
    const int rb  = (lane & 16) ? 8 : 0;
    const float b2 = B2[col];
#pragma unroll
    for (int r = 0; r < 8; ++r) {
        const long n = nb0 + rb + r;
        const float v = o[r] + b2 + nlf[n * 128 + col];
        nlf[n * 128 + col]  = v;
        nl16[n * 128 + col] = (_Float16)v;
    }
}

// ---------------------------------------------------------------------------
// Decoder: L -> L (WMMA) -> ReLU -> 3 (scalar)
// ---------------------------------------------------------------------------
__global__ __launch_bounds__(256) void k_decoder(
    const _Float16* __restrict__ nl16,
    const v16h* __restrict__ W1, const float* __restrict__ B1,
    const float* __restrict__ W2, const float* __restrict__ B2,
    float* __restrict__ out)
{
    __shared__ _Float16 Ain[16 * 136];
    __shared__ float Hf[16 * 132];
    const int tid = threadIdx.x, lane = tid & 31, wave = tid >> 5;
    const long nb0 = (long)blockIdx.x * 16;

    for (int idx = tid; idx < 16 * 64; idx += 256) {
        const int row = idx >> 6, cp = idx & 63;
        const unsigned* src = (const unsigned*)(nl16 + (nb0 + row) * 128);
        *(unsigned*)(Ain + row * 136 + cp * 2) = src[cp];
    }
    __syncthreads();
    v8f h = wmma_gemm_lds(Ain, 136, W1 + (size_t)wave * 4 * 32, 4, lane);
    {
        const int col = wave * 16 + (lane & 15);
        const int rb  = (lane & 16) ? 8 : 0;
        const float b = B1[col];
#pragma unroll
        for (int r = 0; r < 8; ++r) {
            const float v = h[r] + b;
            Hf[(rb + r) * 132 + col] = v > 0.0f ? v : 0.0f;
        }
    }
    __syncthreads();
    if (tid < 48) {
        const int row = tid / 3, d = tid - 3 * (tid / 3);
        float acc = B2[d];
        for (int k = 0; k < 128; ++k) acc += Hf[row * 132 + k] * W2[k * 3 + d];
        out[(nb0 + row) * 3 + d] = acc;
    }
}

// ---------------------------------------------------------------------------
extern "C" void kernel_launch(void* const* d_in, const int* in_sizes, int n_in,
                              void* d_out, int out_size, void* d_ws, size_t ws_size,
                              hipStream_t stream)
{
    const float* pos   = (const float*)d_in[0];
    const float* vel   = (const float*)d_in[1];
    const int*   rid   = (const int*)d_in[2];
    const int*   snd   = (const int*)d_in[3];
    const int*   rcv   = (const int*)d_in[4];
    const float* remb  = (const float*)d_in[5];
    const float* ne_w1 = (const float*)d_in[6];  const float* ne_b1 = (const float*)d_in[7];
    const float* ne_w2 = (const float*)d_in[8];  const float* ne_b2 = (const float*)d_in[9];
    const float* ee_w1 = (const float*)d_in[10]; const float* ee_b1 = (const float*)d_in[11];
    const float* ee_w2 = (const float*)d_in[12]; const float* ee_b2 = (const float*)d_in[13];
    const float* pe_w1 = (const float*)d_in[14]; const float* pe_b1 = (const float*)d_in[15];
    const float* pe_w2 = (const float*)d_in[16]; const float* pe_b2 = (const float*)d_in[17];
    const float* pn_w1 = (const float*)d_in[18]; const float* pn_b1 = (const float*)d_in[19];
    const float* pn_w2 = (const float*)d_in[20]; const float* pn_b2 = (const float*)d_in[21];
    const float* de_w1 = (const float*)d_in[22]; const float* de_b1 = (const float*)d_in[23];
    const float* de_w2 = (const float*)d_in[24]; const float* de_b2 = (const float*)d_in[25];

    const int N = in_sizes[2];
    const int E = in_sizes[3];
    const int L = 128;
    const int STEPS = in_sizes[15] / L;   // pe_b1 is [STEPS, L]

    // -------- carve workspace --------
    char* p = (char*)d_ws;
    auto carve = [&](size_t bytes) -> char* {
        char* r = p; p += (bytes + 255) & ~(size_t)255; return r;
    };
    float*     deg   = (float*)carve((size_t)N * 4);
    float*     csum  = (float*)carve(4 * 4);
    float*     agg   = (float*)carve((size_t)N * L * 4);
    float*     nlf   = (float*)carve((size_t)N * L * 4);
    _Float16*  nl16  = (_Float16*)carve((size_t)N * L * 2);
    _Float16*  el16  = (_Float16*)carve((size_t)E * L * 2);
    auto fragbytes = [](int KB, int NB) { return (size_t)KB * NB * 512 * 2; };
    v16h* wNE1 = (v16h*)carve(fragbytes(1, 8));
    v16h* wNE2 = (v16h*)carve(fragbytes(4, 8));
    v16h* wEE1 = (v16h*)carve(fragbytes(1, 8));
    v16h* wEE2 = (v16h*)carve(fragbytes(4, 8));
    v16h* wPE1 = (v16h*)carve(fragbytes(12, 8) * STEPS);
    v16h* wPE2 = (v16h*)carve(fragbytes(4, 8) * STEPS);
    v16h* wPN1 = (v16h*)carve(fragbytes(8, 8) * STEPS);
    v16h* wPN2 = (v16h*)carve(fragbytes(4, 8) * STEPS);
    v16h* wDE1 = (v16h*)carve(fragbytes(4, 8));

    // -------- weight conversion to WMMA fragment layout --------
    auto conv = [&](const float* w, v16h* dst, int Ksrc, int KB, int Ncols) {
        const int NB = Ncols / 16;
        const int total = NB * KB * 512;
        k_convert_weight<<<(total + 255) / 256, 256, 0, stream>>>(
            w, (_Float16*)dst, Ksrc, KB, NB, Ncols);
    };
    conv(ne_w1, wNE1, 23, 1, 128);
    conv(ne_w2, wNE2, 128, 4, 128);
    conv(ee_w1, wEE1, 9, 1, 128);
    conv(ee_w2, wEE2, 128, 4, 128);
    for (int s = 0; s < STEPS; ++s) {
        conv(pe_w1 + (size_t)s * 384 * 128, wPE1 + (size_t)s * 12 * 8 * 32, 384, 12, 128);
        conv(pe_w2 + (size_t)s * 128 * 128, wPE2 + (size_t)s * 4 * 8 * 32, 128, 4, 128);
        conv(pn_w1 + (size_t)s * 256 * 128, wPN1 + (size_t)s * 8 * 8 * 32, 256, 8, 128);
        conv(pn_w2 + (size_t)s * 128 * 128, wPN2 + (size_t)s * 4 * 8 * 32, 128, 4, 128);
    }
    conv(de_w1, wDE1, 128, 4, 128);

    // -------- graph stats --------
    k_zero<<<256, 256, 0, stream>>>(deg, (long)N);
    k_zero<<<1, 64, 0, stream>>>(csum, 4);
    k_degree<<<256, 256, 0, stream>>>(rcv, deg, E);
    k_center<<<64, 256, 0, stream>>>(pos, csum, N);

    // -------- encoders --------
    k_node_encoder<<<N / 16, 256, 0, stream>>>(
        pos, vel, rid, remb, deg, csum, nlf, nl16,
        wNE1, ne_b1, wNE2, ne_b2, 1.0f / (float)N);
    k_edge_encoder<<<E / 16, 256, 0, stream>>>(
        pos, vel, rid, snd, rcv, el16, wEE1, ee_b1, wEE2, ee_b2);

    // -------- processor --------
    for (int s = 0; s < STEPS; ++s) {
        k_zero<<<1024, 256, 0, stream>>>(agg, (long)N * L);
        k_edge_mlp<<<E / 32, 256, 0, stream>>>(
            nl16, el16, agg, snd, rcv,
            wPE1 + (size_t)s * 12 * 8 * 32, pe_b1 + (size_t)s * L,
            wPE2 + (size_t)s * 4 * 8 * 32,  pe_b2 + (size_t)s * L);
        k_node_mlp<<<N / 16, 256, 0, stream>>>(
            nlf, nl16, agg,
            wPN1 + (size_t)s * 8 * 8 * 32, pn_b1 + (size_t)s * L,
            wPN2 + (size_t)s * 4 * 8 * 32, pn_b2 + (size_t)s * L);
    }

    // -------- decoder --------
    k_decoder<<<N / 16, 256, 0, stream>>>(nl16, wDE1, de_b1, de_w2, de_b2,
                                          (float*)d_out);
}
